// hFusedLiveSR_2559800508449
// MI455X (gfx1250) — matrix operand order repaired and checked by reading, compiled, not run on hardware
//
#include <hip/hip_runtime.h>
#include <hip/hip_bf16.h>

// ---------------------------------------------------------------------------
// MoE super-resolution (4-expert EDSR-lite), CDNA5 / gfx1250, wave32.
// Tap-decomposed implicit GEMM: per 3x3 tap, one v_wmma_f32_16x16x32_f16 with
// K = Cin = 32. Channel-last fp16 activations; input tile staged to LDS with
// zero-padded halo; weights pre-repacked to [e][tap][cout][cin] fp16 so A
// fragments are contiguous b128 loads at compile-time immediate offsets;
// biases pre-repacked to zero-padded [e][COUTPAD] f32 so the epilogue is two
// unguarded v4f loads. Per Cout-tile: 9 A-fragment load pairs software-
// pipelined against 9 chained WMMAs (B fragments register-resident).
// ---------------------------------------------------------------------------

typedef __attribute__((ext_vector_type(16))) _Float16 v16h;
typedef __attribute__((ext_vector_type(8)))  _Float16 v8h;
typedef __attribute__((ext_vector_type(8)))  float    v8f;
typedef __attribute__((ext_vector_type(4)))  float    v4f;

#define N_IMG   128
#define E_EXP   4
#define CAP     48
#define NSLOT   (E_EXP * CAP)   // 192
#define NB      4
#define F_CH    32
#define RES_SCALE 0.1f

// conv epilogue / input modes (template parameter MODE)
#define M_RELU 1
#define M_ADD  2   // out = addsrc + scale*(acc+bias); addsrc channel-last C=32
#define M_PSIN 4   // input buffer is raw 128-ch half-res (pixel-shuffle read)
#define M_TAIL 8   // final conv: scatter 3 f32 channels into d_out via pos[]

// ---------------------------------------------------------------------------
// Classifier: conv3->32 (SAME) + relu + GAP + linear 32->4 + argmax.
// ---------------------------------------------------------------------------
__global__ void cls_kernel(const float* __restrict__ inp,
                           const float* __restrict__ w,
                           const float* __restrict__ b,
                           const float* __restrict__ fw,
                           const float* __restrict__ fb,
                           int* __restrict__ assign) {
  __shared__ float ssum[32];
  const int n = blockIdx.x;
  const int t = threadIdx.x;
  if (t < 32) ssum[t] = 0.f;
  __syncthreads();

  float local[32];
  #pragma unroll
  for (int o = 0; o < 32; ++o) local[o] = 0.f;

  for (int p = t; p < 1024; p += 256) {
    const int y = p >> 5, x = p & 31;
    float patch[27];
    #pragma unroll
    for (int i = 0; i < 3; ++i)
      #pragma unroll
      for (int ky = 0; ky < 3; ++ky)
        #pragma unroll
        for (int kx = 0; kx < 3; ++kx) {
          const int yy = y + ky - 1, xx = x + kx - 1;
          patch[(i * 3 + ky) * 3 + kx] =
              (yy >= 0 && yy < 32 && xx >= 0 && xx < 32)
                  ? inp[(((size_t)n * 3 + i) * 32 + yy) * 32 + xx] : 0.f;
        }
    for (int o = 0; o < 32; ++o) {
      float a = b[o];
      #pragma unroll
      for (int k = 0; k < 27; ++k) a += patch[k] * w[o * 27 + k];
      local[o] += fmaxf(a, 0.f);
    }
  }
  #pragma unroll
  for (int o = 0; o < 32; ++o) atomicAdd(&ssum[o], local[o]);
  __syncthreads();

  if (t == 0) {
    int best = 0; float bs = -3.0e38f;
    for (int e = 0; e < E_EXP; ++e) {
      float s = fb[e];
      for (int o = 0; o < 32; ++o)
        s += (ssum[o] * (1.f / 1024.f)) * fw[e * 32 + o];
      if (s > bs) { bs = s; best = e; }   // strict >: first max wins
    }
    assign[n] = best;
  }
}

// per-expert sorted position lists padded to CAP (sentinel N_IMG)
__global__ void route_kernel(const int* __restrict__ assign,
                             int* __restrict__ pos) {
  const int e = threadIdx.x;
  if (e >= E_EXP) return;
  int cnt = 0;
  for (int n = 0; n < N_IMG; ++n)
    if (assign[n] == e && cnt < CAP) pos[e * CAP + cnt++] = n;
  for (; cnt < CAP; ++cnt) pos[e * CAP + cnt] = N_IMG;
}

// ---------------------------------------------------------------------------
// Gather inputs (NCHW f32) -> channel-last fp16 [NSLOT][32][32][32] (C padded
// 3->32 with zeros); invalid slots fully zeroed.
// ---------------------------------------------------------------------------
__global__ void gather_kernel(const float* __restrict__ inp,
                              const int* __restrict__ pos,
                              _Float16* __restrict__ xbuf) {
  const int img = blockIdx.x;
  const int n = pos[img];
  for (int p = threadIdx.x; p < 1024; p += 256) {
    union { v8h v[4]; _Float16 h[32]; } t;
    #pragma unroll
    for (int c = 0; c < 32; ++c) t.h[c] = (_Float16)0.f;
    if (n < N_IMG) {
      #pragma unroll
      for (int c = 0; c < 3; ++c)
        t.h[c] = (_Float16)inp[((size_t)n * 3 + c) * 1024 + p];
    }
    v8h* dst = (v8h*)(xbuf + ((size_t)img * 1024 + p) * 32);
    dst[0] = t.v[0]; dst[1] = t.v[1]; dst[2] = t.v[2]; dst[3] = t.v[3];
  }
}

__global__ void zero_f32_kernel(float* __restrict__ p, size_t n) {
  size_t i = (size_t)blockIdx.x * blockDim.x + threadIdx.x;
  size_t stride = (size_t)gridDim.x * blockDim.x;
  for (; i < n; i += stride) p[i] = 0.f;
}

// ---------------------------------------------------------------------------
// Weight repack: f32 [E][Cout][Cin][3][3] -> fp16 [E][9][CoutPad][32]
// (cin padded to 32, cout padded to CoutPad with zeros).
// grid(E, CoutPad), block(288) = (tap, cin).
// ---------------------------------------------------------------------------
__global__ void repack_w_kernel(const float* __restrict__ src,
                                _Float16* __restrict__ dst,
                                int Cout, int Cin, int CoutPad,
                                long estride /* src per-expert elements */) {
  const int e = blockIdx.x, cout = blockIdx.y;
  const int tap = threadIdx.x / 32, cin = threadIdx.x % 32;
  float v = 0.f;
  if (cout < Cout && cin < Cin)
    v = src[(size_t)e * estride + ((size_t)cout * Cin + cin) * 9 + tap];
  dst[(((size_t)e * 9 + tap) * CoutPad + cout) * 32 + cin] = (_Float16)v;
}

// Bias repack: f32 [E][CoutReal] -> f32 [E][CoutPad] zero-padded.
__global__ void repack_b_kernel(const float* __restrict__ src,
                                float* __restrict__ dst,
                                int CoutReal, int CoutPad, long estride) {
  const int e = blockIdx.x, c = threadIdx.x;
  dst[(size_t)e * CoutPad + c] =
      (c < CoutReal) ? src[(size_t)e * estride + c] : 0.f;
}

// ---------------------------------------------------------------------------
// Tap-decomposed 3x3 SAME conv, Cin=32, via WMMA f32<=f16 16x16x32.
//  block = 128 threads (4 waves); pixel tile = 4 rows x 16 cols, wave w = row.
//  grid = (NSLOT, H/4, W/16). Each wave loops over COUTPAD/16 tiles.
//  LDS: 6x18x32 fp16 halo'd input slab, staged once (zero-padded borders).
//  Per Cout-tile: 9 A-load pairs pipelined against 9 chained WMMAs.
// ---------------------------------------------------------------------------
template <int COUTPAD, int MODE>
__global__ __launch_bounds__(128)
void conv_tap_wmma(const _Float16* __restrict__ in,
                   _Float16* __restrict__ out,          // [img][H][W][COUTPAD]
                   const _Float16* __restrict__ addsrc, // [img][H][W][32]
                   const _Float16* __restrict__ wrep,   // [E][9][COUTPAD][32]
                   const float* __restrict__ bias,      // [E][COUTPAD] padded
                   const int* __restrict__ pos,
                   float* __restrict__ finalout,
                   int H, int W, float scale) {
  __shared__ __align__(16) _Float16 slab[6 * 18 * 32];

  const int tid  = threadIdx.x;
  const int lane = tid & 31;
  const int wv   = tid >> 5;
  const int img  = blockIdx.x;
  const int e    = img / CAP;
  const int y0   = blockIdx.y * 4;
  const int x0   = blockIdx.z * 16;

  // ---- stage halo'd input tile: rows y0-1..y0+4, cols x0-1..x0+16 ----
  if (tid < 108) {
    const int ly = tid / 18, lx = tid - ly * 18;
    const int gy = y0 - 1 + ly, gx = x0 - 1 + lx;
    union { v8h v[4]; _Float16 h[32]; } t;
    if (gy >= 0 && gy < H && gx >= 0 && gx < W) {
      if (MODE & M_PSIN) {
        // raw buffer: [img][H/2][W/2][128]; logical (gy,gx,c) reads raw
        // channel c*4 + (gy&1)*2 + (gx&1)
        const _Float16* src = in +
            (((size_t)img * (H >> 1) + (gy >> 1)) * (W >> 1) + (gx >> 1)) * 128 +
            ((gy & 1) << 1) + (gx & 1);
        #pragma unroll
        for (int c = 0; c < 32; ++c) t.h[c] = src[c * 4];
      } else {
        const v8h* src = (const v8h*)(in + (((size_t)img * H + gy) * W + gx) * 32);
        t.v[0] = src[0]; t.v[1] = src[1]; t.v[2] = src[2]; t.v[3] = src[3];
      }
    } else {
      #pragma unroll
      for (int c = 0; c < 32; ++c) t.h[c] = (_Float16)0.f;
    }
    v8h* d = (v8h*)(slab + ((size_t)ly * 18 + lx) * 32);
    d[0] = t.v[0]; d[1] = t.v[1]; d[2] = t.v[2]; d[3] = t.v[3];
  }
  __syncthreads();

  const int hi = lane >> 4;      // lane-half selector
  const int ln = lane & 15;
  const int y  = y0 + wv;
  const int x  = x0 + ln;

  // ---- B fragments for all 9 taps, register-resident ----
  // B cols: lane ln = pixel (y, x); K halves = 16 contiguous cin at 16*hi
  union BF { v16h v; v8h p[2]; };
  BF Bf[9];
  #pragma unroll
  for (int tap = 0; tap < 9; ++tap) {
    const int ky = tap / 3, kx = tap % 3;
    const _Float16* bp = slab + (((wv + ky) * 18) + (ln + kx)) * 32 + 16 * hi;
    Bf[tap].p[0] = *(const v8h*)(bp);
    Bf[tap].p[1] = *(const v8h*)(bp + 8);
  }

  const _Float16* wbase = wrep + (size_t)e * 9 * COUTPAD * 32;
  constexpr int NCT = COUTPAD >> 4;

  #pragma unroll
  for (int ct = 0; ct < NCT; ++ct) {
    // A rows: lane ln owns cout = ct*16 + ln; K halves = 16 contiguous cin at 16*hi.
    const _Float16* wct = wbase + ((size_t)ct * 16 + ln) * 32 + 16 * hi;
    __builtin_prefetch(wct, 0, 3);
    BF Af[9];
    #pragma unroll
    for (int tap = 0; tap < 9; ++tap) {
      const _Float16* wp = wct + tap * (COUTPAD * 32);  // immediate offsets
      Af[tap].p[0] = *(const v8h*)(wp);
      Af[tap].p[1] = *(const v8h*)(wp + 8);
    }
    v8f acc = {};
    #pragma unroll
    for (int tap = 0; tap < 9; ++tap)
      acc = __builtin_amdgcn_wmma_f32_16x16x32_f16(
          false, Af[tap].v, false, Bf[tap].v, (short)0, acc, false, false);

    // ---- epilogue: lane holds 8 contiguous couts (cbase..cbase+7) at (y,x) --
    union { v8f v; float f[8]; } C; C.v = acc;
    const int cbase = ct * 16 + 8 * hi;
    // zero-padded bias: two unguarded 16B loads
    union { v4f v[2]; float f[8]; } bb;
    const float* be = bias + (size_t)e * COUTPAD + cbase;
    bb.v[0] = *(const v4f*)(be);
    bb.v[1] = *(const v4f*)(be + 4);
    float vals[8];
    #pragma unroll
    for (int r = 0; r < 8; ++r) {
      float v = C.f[r] + bb.f[r];
      if (MODE & M_RELU) v = fmaxf(v, 0.f);
      vals[r] = v;
    }
    if (MODE & M_ADD) {
      union { v8h v; _Float16 h[8]; } a;
      a.v = *(const v8h*)(addsrc + (((size_t)img * H + y) * W + x) * 32 + cbase);
      #pragma unroll
      for (int r = 0; r < 8; ++r) vals[r] = (float)a.h[r] + scale * vals[r];
    }
    if (MODE & M_TAIL) {
      // COUTPAD==16, single ct; real channels are cout 0..2 -> hi==0, r<3
      const int nimg = pos[img];
      if (nimg < N_IMG && hi == 0) {
        #pragma unroll
        for (int r = 0; r < 3; ++r)
          finalout[(((size_t)nimg * 3 + r) * H + y) * W + x] = vals[r];
      }
    } else {
      union { v8h v; _Float16 h[8]; } o;
      #pragma unroll
      for (int r = 0; r < 8; ++r) o.h[r] = (_Float16)vals[r];
      *(v8h*)(out + (((size_t)img * H + y) * W + x) * COUTPAD + cbase) = o.v;
    }
  }
}

// ---------------------------------------------------------------------------
// Host-side orchestration
// ---------------------------------------------------------------------------
extern "C" void kernel_launch(void* const* d_in, const int* in_sizes, int n_in,
                              void* d_out, int out_size, void* d_ws, size_t ws_size,
                              hipStream_t stream) {
  const float* inputs = (const float*)d_in[0];
  const float* cls_w  = (const float*)d_in[1];
  const float* cls_b  = (const float*)d_in[2];
  const float* cls_fw = (const float*)d_in[3];
  const float* cls_fb = (const float*)d_in[4];
  const float* head_w = (const float*)d_in[5];
  const float* head_b = (const float*)d_in[6];
  const float* rb_w1  = (const float*)d_in[7];
  const float* rb_b1  = (const float*)d_in[8];
  const float* rb_w2  = (const float*)d_in[9];
  const float* rb_b2  = (const float*)d_in[10];
  const float* end_w  = (const float*)d_in[11];
  const float* end_b  = (const float*)d_in[12];
  const float* up_w1  = (const float*)d_in[13];
  const float* up_b1  = (const float*)d_in[14];
  const float* up_w2  = (const float*)d_in[15];
  const float* up_b2  = (const float*)d_in[16];
  const float* tail_w = (const float*)d_in[17];
  const float* tail_b = (const float*)d_in[18];
  float* out = (float*)d_out;
  (void)ws_size; (void)n_in; (void)in_sizes; (void)out_size;

  char* base = (char*)d_ws;
  size_t off = 0;
  auto carve = [&](size_t bytes) -> char* {
    char* p = base + off;
    off = (off + bytes + 255) & ~(size_t)255;
    return p;
  };
  int*      assign = (int*)     carve(N_IMG * sizeof(int));
  int*      pos    = (int*)     carve(NSLOT * sizeof(int));
  _Float16* xbuf   = (_Float16*)carve((size_t)NSLOT * 1024 * 32 * 2);
  _Float16* hbuf   = (_Float16*)carve((size_t)NSLOT * 1024 * 32 * 2);
  _Float16* r0     = (_Float16*)carve((size_t)NSLOT * 1024 * 32 * 2);
  _Float16* r1     = (_Float16*)carve((size_t)NSLOT * 1024 * 32 * 2);
  _Float16* tbuf   = (_Float16*)carve((size_t)NSLOT * 1024 * 32 * 2);
  _Float16* u1raw  = (_Float16*)carve((size_t)NSLOT * 1024 * 128 * 2);  // 32x32x128
  _Float16* u2raw  = (_Float16*)carve((size_t)NSLOT * 4096 * 128 * 2);  // 64x64x128
  // repacked weights (fp16, [e][tap][coutPad][32])
  const size_t W32 = (size_t)E_EXP * 9 * 32 * 32;    // CoutPad 32
  const size_t W128 = (size_t)E_EXP * 9 * 128 * 32;  // CoutPad 128
  const size_t W16 = (size_t)E_EXP * 9 * 16 * 32;    // CoutPad 16
  _Float16* wr_head = (_Float16*)carve(W32 * 2);
  _Float16* wr_rb1  = (_Float16*)carve(NB * W32 * 2);
  _Float16* wr_rb2  = (_Float16*)carve(NB * W32 * 2);
  _Float16* wr_end  = (_Float16*)carve(W32 * 2);
  _Float16* wr_up1  = (_Float16*)carve(W128 * 2);
  _Float16* wr_up2  = (_Float16*)carve(W128 * 2);
  _Float16* wr_tail = (_Float16*)carve(W16 * 2);
  // repacked biases (f32, [e][coutPad], zero-padded)
  float* br_head = (float*)carve(E_EXP * 32 * 4);
  float* br_rb1  = (float*)carve(NB * E_EXP * 32 * 4);
  float* br_rb2  = (float*)carve(NB * E_EXP * 32 * 4);
  float* br_end  = (float*)carve(E_EXP * 32 * 4);
  float* br_up1  = (float*)carve(E_EXP * 128 * 4);
  float* br_up2  = (float*)carve(E_EXP * 128 * 4);
  float* br_tail = (float*)carve(E_EXP * 16 * 4);

  // 0) zero output (tail scatters only valid rows)
  zero_f32_kernel<<<2048, 256, 0, stream>>>(out, (size_t)N_IMG * 3 * 128 * 128);

  // 1) classify + route + gather (channel-padded, channel-last fp16)
  cls_kernel<<<N_IMG, 256, 0, stream>>>(inputs, cls_w, cls_b, cls_fw, cls_fb, assign);
  route_kernel<<<1, 32, 0, stream>>>(assign, pos);
  gather_kernel<<<NSLOT, 256, 0, stream>>>(inputs, pos, xbuf);

  // 2) repack all weights/biases
  repack_w_kernel<<<dim3(E_EXP, 32), 288, 0, stream>>>(head_w, wr_head, 32, 3, 32, 32 * 27);
  repack_b_kernel<<<E_EXP, 32, 0, stream>>>(head_b, br_head, 32, 32, 32);
  for (int i = 0; i < NB; ++i) {
    repack_w_kernel<<<dim3(E_EXP, 32), 288, 0, stream>>>(
        rb_w1 + (size_t)i * F_CH * F_CH * 9, wr_rb1 + (size_t)i * W32,
        32, 32, 32, (long)NB * F_CH * F_CH * 9);
    repack_w_kernel<<<dim3(E_EXP, 32), 288, 0, stream>>>(
        rb_w2 + (size_t)i * F_CH * F_CH * 9, wr_rb2 + (size_t)i * W32,
        32, 32, 32, (long)NB * F_CH * F_CH * 9);
    repack_b_kernel<<<E_EXP, 32, 0, stream>>>(
        rb_b1 + (size_t)i * F_CH, br_rb1 + (size_t)i * E_EXP * 32, 32, 32, NB * F_CH);
    repack_b_kernel<<<E_EXP, 32, 0, stream>>>(
        rb_b2 + (size_t)i * F_CH, br_rb2 + (size_t)i * E_EXP * 32, 32, 32, NB * F_CH);
  }
  repack_w_kernel<<<dim3(E_EXP, 32), 288, 0, stream>>>(end_w, wr_end, 32, 32, 32, 32 * 288);
  repack_b_kernel<<<E_EXP, 32, 0, stream>>>(end_b, br_end, 32, 32, 32);
  repack_w_kernel<<<dim3(E_EXP, 128), 288, 0, stream>>>(up_w1, wr_up1, 128, 32, 128, 128 * 288);
  repack_b_kernel<<<E_EXP, 128, 0, stream>>>(up_b1, br_up1, 128, 128, 128);
  repack_w_kernel<<<dim3(E_EXP, 128), 288, 0, stream>>>(up_w2, wr_up2, 128, 32, 128, 128 * 288);
  repack_b_kernel<<<E_EXP, 128, 0, stream>>>(up_b2, br_up2, 128, 128, 128);
  repack_w_kernel<<<dim3(E_EXP, 16), 288, 0, stream>>>(tail_w, wr_tail, 3, 32, 16, 3 * 288);
  repack_b_kernel<<<E_EXP, 16, 0, stream>>>(tail_b, br_tail, 3, 16, 3);

  // 3) head: 3(->32) -> 32 @32x32
  conv_tap_wmma<32, 0><<<dim3(NSLOT, 8, 2), 128, 0, stream>>>(
      xbuf, hbuf, nullptr, wr_head, br_head, nullptr, nullptr, 32, 32, 0.f);

  // 4) 4 ResBlocks: t = relu(conv1(cur)); next = cur + 0.1*conv2(t)
  _Float16* cur = hbuf;
  for (int i = 0; i < NB; ++i) {
    _Float16* next = (i & 1) ? r1 : r0;
    conv_tap_wmma<32, M_RELU><<<dim3(NSLOT, 8, 2), 128, 0, stream>>>(
        cur, tbuf, nullptr, wr_rb1 + (size_t)i * W32,
        br_rb1 + (size_t)i * E_EXP * 32, nullptr, nullptr, 32, 32, 0.f);
    conv_tap_wmma<32, M_ADD><<<dim3(NSLOT, 8, 2), 128, 0, stream>>>(
        tbuf, next, cur, wr_rb2 + (size_t)i * W32,
        br_rb2 + (size_t)i * E_EXP * 32, nullptr, nullptr, 32, 32, RES_SCALE);
    cur = next;
  }

  // 5) body end + skip(h)
  _Float16* rend = (cur == r0) ? r1 : r0;
  conv_tap_wmma<32, M_ADD><<<dim3(NSLOT, 8, 2), 128, 0, stream>>>(
      cur, rend, hbuf, wr_end, br_end, nullptr, nullptr, 32, 32, 1.f);

  // 6) up1: 32 -> 128 @32x32 (raw store; shuffle folded into consumer)
  conv_tap_wmma<128, 0><<<dim3(NSLOT, 8, 2), 128, 0, stream>>>(
      rend, u1raw, nullptr, wr_up1, br_up1, nullptr, nullptr, 32, 32, 0.f);

  // 7) up2: logical 64x64x32 input read through shuffle of u1raw -> 128 raw
  conv_tap_wmma<128, M_PSIN><<<dim3(NSLOT, 16, 4), 128, 0, stream>>>(
      u1raw, u2raw, nullptr, wr_up2, br_up2, nullptr, nullptr, 64, 64, 0.f);

  // 8) tail: logical 128x128x32 input through shuffle of u2raw -> 3 ch f32,
  //    scattered into d_out at pos[]
  conv_tap_wmma<16, M_PSIN | M_TAIL><<<dim3(NSLOT, 32, 8), 128, 0, stream>>>(
      u2raw, nullptr, nullptr, wr_tail, br_tail, pos, out, 128, 128, 0.f);
}
